// TDLambdaLoss_68298569941724
// MI455X (gfx1250) — matrix operand order repaired and checked by reading, not compiled, not run on hardware
//
#include <hip/hip_runtime.h>
#include <hip/hip_bf16.h>
#include <math.h>

// ---------------------------------------------------------------------------
// TD(lambda) distributional value loss for MI455X (gfx1250, wave32, WMMA).
// Dominant cost: (17408 x 4096) @ (4096 x 255) GEMM -> bf16 WMMA w/ f32 acc.
// Latents streamed once via GLOBAL_LOAD_ASYNC_TO_LDS (ASYNCcnt) so the HBM
// stream is decoupled from the B-fragment LOADcnt waits; B software-pipelined
// one k-step ahead in two register banks. Packed W (2MB) stays L2-resident.
// ---------------------------------------------------------------------------

#define BATCH   1024
#define HORIZON 16
#define DIM     4096
#define NBINS   255
#define KPAD    256          // padded bin dim (16 n-tiles of 16)
#define NROWS   (BATCH*HORIZON)      // 16384 main rows
#define ROWS_PB 32                   // rows per block (2 WMMA M-tiles)
#define NBLK_MAIN (NROWS/ROWS_PB)    // 512
#define NBLK_BOOT (BATCH/ROWS_PB)    // 32
#define NBLK    (NBLK_MAIN + NBLK_BOOT)
#define NKT     (DIM/32)             // 128 k-tiles

#define DISCOUNT 0.997f
#define LAMBDA   0.95f
#define LOSS_SCALE 0.5f
#define BIN_LO (-20.0f)
#define BIN_STEP (40.0f/254.0f)

typedef __attribute__((ext_vector_type(16))) __bf16 v16bf;
typedef __attribute__((ext_vector_type(8)))  float  v8f;
typedef __attribute__((ext_vector_type(4)))  float  v4f;
typedef __attribute__((ext_vector_type(4)))  unsigned int v4u;

__device__ __forceinline__ unsigned int f2bf(float f) {
    // round-to-nearest-even f32 -> bf16 (integer path, guaranteed lowering)
    unsigned int u = __float_as_uint(f);
    u += 0x7fffu + ((u >> 16) & 1u);
    return u >> 16;
}

__device__ __forceinline__ unsigned int pk2bf(float lo, float hi) {
    return f2bf(lo) | (f2bf(hi) << 16);
}

// ---------------------------------------------------------------------------
// Kernel 1: pack W (4096 x 255 f32) into bf16 arranged in the exact WMMA
// B-fragment layout: index = ((kt*16 + nt)*32 + lane)*16 + e
//   lane = N (mod 16); table order: e<8 -> K=kt*32+8h+e, e>=8 -> K=kt*32+16+8h+(e-8)
//   element e packs two-per-VGPR (low=even K), h = lane/16.
// Column 255 is zero padding. 1M elements, one per thread.
// ---------------------------------------------------------------------------
__global__ __launch_bounds__(256) void pack_w_kernel(const float* __restrict__ W,
                                                     unsigned short* __restrict__ packB) {
    unsigned g = blockIdx.x * 256u + threadIdx.x;          // [0, 4096*256)
    unsigned e    = g & 15u;
    unsigned lane = (g >> 4) & 31u;
    unsigned nt   = (g >> 9) & 15u;
    unsigned kt   = g >> 13;
    unsigned K    = kt * 32u + ((e & 8u) ? 16u : 0u) + ((lane & 16u) ? 8u : 0u) + (e & 7u);
    unsigned col  = nt * 16u + (lane & 15u);
    float val = (col < NBINS) ? W[K * NBINS + col] : 0.0f;
    packB[g] = (unsigned short)f2bf(val);
}

// ---------------------------------------------------------------------------
// Kernel 2: GEMM + value head. 1 block (4 wave32) per 32-row slab (2 M-tiles).
// ---------------------------------------------------------------------------
__global__ __launch_bounds__(128) void gemm_head_kernel(
        const float* __restrict__ latents,
        const float* __restrict__ boot,
        const float* __restrict__ bias,
        const unsigned short* __restrict__ packB,
        float* __restrict__ logits,     // [NROWS][KPAD]
        float* __restrict__ values,     // [NROWS]
        float* __restrict__ lse,        // [NROWS]
        float* __restrict__ bootval)    // [BATCH]
{
    __shared__ float rawA[2][128][32];               // raw f32 async landing, 32KB
    __shared__ unsigned short AchB[2][2][4][32][16]; // [buf][mt][ks][lane][e] 16KB
    __shared__ float Ltile[ROWS_PB][KPAD];           // 32KB epilogue tile
    __shared__ float Pa[ROWS_PB][4], Pb[ROWS_PB][4], rowmax32[ROWS_PB];

    const int blk = blockIdx.x;
    const bool isBoot = (blk >= NBLK_MAIN);
    const float* src = isBoot ? (boot + (size_t)(blk - NBLK_MAIN) * ROWS_PB * DIM)
                              : (latents + (size_t)blk * ROWS_PB * DIM);

    const int tid  = threadIdx.x;
    const int lane = tid & 31;
    const int wave = tid >> 5;

    v8f acc[2][4] = {};

    // Thread t owns rows {lm, lm+16} x 16 consecutive K at c0=(t%8)*16.
    //   kk = c0 & 31 is 0 or 16:
    //   kk 0..7   -> (lane=lm,    e=0..7)   | kk 8..15  -> (lane=lm+16, e=0..7)
    //   kk 16..23 -> (lane=lm,    e=8..15)  | kk 24..31 -> (lane=lm+16, e=8..15)
    const int lm  = tid >> 3;
    const int c0  = (tid & 7) * 16;
    const int ksL = c0 >> 5;
    const int e0  = (c0 & 16) ? 8 : 0;

    // Async-copy this thread's 32 floats of chunk kcn into rawA[buf][tid][0..31]
    // (floats 0..15 = row lm, 16..31 = row lm+16). ASYNCcnt-tracked: decoupled
    // from the B-fragment LOADcnt waits in the WMMA loop.
    auto async_issue = [&](int buf, int kcn) {
        const float* g0 = src + (size_t)lm * DIM + kcn + c0;
        const float* g1 = src + (size_t)(lm + 16) * DIM + kcn + c0;
        unsigned l0 = (unsigned)(unsigned long long)&rawA[buf][tid][0];
        #pragma unroll
        for (int q = 0; q < 4; ++q)
            asm volatile("global_load_async_to_lds_b128 %0, %1, off th:TH_LOAD_NT"
                         :: "v"(l0 + q * 16),
                            "v"((unsigned long long)(g0 + q * 4)) : "memory");
        #pragma unroll
        for (int q = 0; q < 4; ++q)
            asm volatile("global_load_async_to_lds_b128 %0, %1, off th:TH_LOAD_NT"
                         :: "v"(l0 + 64 + q * 16),
                            "v"((unsigned long long)(g1 + q * 4)) : "memory");
    };

    // Convert this thread's own raw region -> bf16 A-fragment layout in LDS.
    auto convert = [&](int buf) {
        const float* rp = &rawA[buf][tid][0];
        v4f p[8];
        #pragma unroll
        for (int q = 0; q < 8; ++q) p[q] = *(const v4f*)(rp + q * 4);
        v4u g;
        g.x = pk2bf(p[0].x, p[0].y);  g.y = pk2bf(p[0].z, p[0].w);
        g.z = pk2bf(p[1].x, p[1].y);  g.w = pk2bf(p[1].z, p[1].w);
        *(v4u*)&AchB[buf][0][ksL][lm][e0] = g;
        g.x = pk2bf(p[2].x, p[2].y);  g.y = pk2bf(p[2].z, p[2].w);
        g.z = pk2bf(p[3].x, p[3].y);  g.w = pk2bf(p[3].z, p[3].w);
        *(v4u*)&AchB[buf][0][ksL][lm + 16][e0] = g;
        g.x = pk2bf(p[4].x, p[4].y);  g.y = pk2bf(p[4].z, p[4].w);
        g.z = pk2bf(p[5].x, p[5].y);  g.w = pk2bf(p[5].z, p[5].w);
        *(v4u*)&AchB[buf][1][ksL][lm][e0] = g;
        g.x = pk2bf(p[6].x, p[6].y);  g.y = pk2bf(p[6].z, p[6].w);
        g.z = pk2bf(p[7].x, p[7].y);  g.w = pk2bf(p[7].z, p[7].w);
        *(v4u*)&AchB[buf][1][ksL][lm + 16][e0] = g;
    };

    // Two-bank software pipeline for B fragments (LOADcnt waits become "<= one
    // bank outstanding" instead of 0).
    v16bf bk[2][4];
    auto loadB = [&](int bank, int kt) {
        const unsigned short* bp =
            packB + ((size_t)(kt * 16 + wave * 4) * 32 + lane) * 16;
        bk[bank][0] = *(const v16bf*)(bp);
        bk[bank][1] = *(const v16bf*)(bp + 512);
        bk[bank][2] = *(const v16bf*)(bp + 1024);
        bk[bank][3] = *(const v16bf*)(bp + 1536);
    };

    // Prologue: stage chunk 0 + first B bank.
    async_issue(0, 0);
    loadB(0, 0);
    asm volatile("s_wait_asynccnt 0x0" ::: "memory");
    convert(0);
    __syncthreads();

    for (int kc = 0; kc < DIM; kc += 128) {
        const int buf = (kc >> 7) & 1;
        const bool more = (kc + 128) < DIM;
        if (more) async_issue(buf ^ 1, kc + 128);   // lands during 32 WMMAs
        #pragma unroll
        for (int ks = 0; ks < 4; ++ks) {
            const int kt = (kc >> 5) + ks;
            if (kt + 1 < NKT) loadB((ks + 1) & 1, kt + 1);   // pipeline B
            v16bf a0 = *(const v16bf*)&AchB[buf][0][ks][lane][0];
            v16bf a1 = *(const v16bf*)&AchB[buf][1][ks][lane][0];
            const v16bf* bc = bk[ks & 1];
            acc[0][0] = __builtin_amdgcn_wmma_f32_16x16x32_bf16(false, a0, false, bc[0], (short)0, acc[0][0], false, false);
            acc[1][0] = __builtin_amdgcn_wmma_f32_16x16x32_bf16(false, a1, false, bc[0], (short)0, acc[1][0], false, false);
            acc[0][1] = __builtin_amdgcn_wmma_f32_16x16x32_bf16(false, a0, false, bc[1], (short)0, acc[0][1], false, false);
            acc[1][1] = __builtin_amdgcn_wmma_f32_16x16x32_bf16(false, a1, false, bc[1], (short)0, acc[1][1], false, false);
            acc[0][2] = __builtin_amdgcn_wmma_f32_16x16x32_bf16(false, a0, false, bc[2], (short)0, acc[0][2], false, false);
            acc[1][2] = __builtin_amdgcn_wmma_f32_16x16x32_bf16(false, a1, false, bc[2], (short)0, acc[1][2], false, false);
            acc[0][3] = __builtin_amdgcn_wmma_f32_16x16x32_bf16(false, a0, false, bc[3], (short)0, acc[0][3], false, false);
            acc[1][3] = __builtin_amdgcn_wmma_f32_16x16x32_bf16(false, a1, false, bc[3], (short)0, acc[1][3], false, false);
        }
        if (more) {
            asm volatile("s_wait_asynccnt 0x0" ::: "memory");  // own region landed
            convert(buf ^ 1);   // other buffer: race-free vs this chunk's readers
        }
        __syncthreads();
    }

    // Epilogue: C layout -> VGPR r holds M=r (lanes0-15) / M=r+8 (lanes16-31)
    {
        const int mbase = (lane & 16) ? 8 : 0;
        const int coll  = lane & 15;
        #pragma unroll
        for (int mt = 0; mt < 2; ++mt) {
            #pragma unroll
            for (int j = 0; j < 4; ++j) {
                const int col = wave * 64 + j * 16 + coll;
                const float bb = (col < NBINS) ? bias[col] : 0.0f;
                #pragma unroll
                for (int r = 0; r < 8; ++r)
                    Ltile[mt * 16 + mbase + r][col] = acc[mt][j][r] + bb;
            }
        }
    }
    __syncthreads();

    // Per-row softmax stats, 4 threads per row (deterministic fixed-order).
    const int rrow = tid & 31;
    const int seg  = tid >> 5;              // 0..3, 64 columns each
    const int cbeg = seg * 64;
    const int cend = (cbeg + 64 < NBINS) ? cbeg + 64 : NBINS;
    {
        float pm = -1e30f;
        for (int c = cbeg; c < cend; ++c) pm = fmaxf(pm, Ltile[rrow][c]);
        Pa[rrow][seg] = pm;
    }
    __syncthreads();
    if (tid < ROWS_PB) {
        float m = Pa[tid][0];
        #pragma unroll
        for (int q = 1; q < 4; ++q) m = fmaxf(m, Pa[tid][q]);
        rowmax32[tid] = m;
    }
    __syncthreads();
    {
        const float rm = rowmax32[rrow];
        float ssum = 0.0f, wsum = 0.0f;
        for (int c = cbeg; c < cend; ++c) {
            float e = __expf(Ltile[rrow][c] - rm);
            ssum += e;
            wsum += e * (BIN_LO + BIN_STEP * (float)c);
        }
        Pa[rrow][seg] = ssum;
        Pb[rrow][seg] = wsum;
    }
    __syncthreads();
    if (tid < ROWS_PB) {
        float s = 0.0f, w = 0.0f;
        #pragma unroll
        for (int q = 0; q < 4; ++q) { s += Pa[tid][q]; w += Pb[tid][q]; }
        const float val = w / s;
        const int gr = blk * ROWS_PB + tid;
        if (gr < NROWS) {
            values[gr] = val;
            lse[gr]    = rowmax32[tid] + __logf(s);
        } else {
            bootval[gr - NROWS] = val;
        }
    }

    // Store logits slab (main rows only) for the later 2-entry two-hot gather.
    if (!isBoot) {
        for (int i = tid; i < ROWS_PB * (KPAD / 4); i += 128) {
            const int r  = i >> 6;
            const int c4 = (i & 63) << 2;
            *(float4*)(logits + ((size_t)(blk * ROWS_PB + r)) * KPAD + c4) =
                *(const float4*)&Ltile[r][c4];
        }
    }
}

// ---------------------------------------------------------------------------
// Kernel 3: backward lambda-return scan, one thread per trajectory.
// ---------------------------------------------------------------------------
__global__ __launch_bounds__(256) void scan_kernel(
        const float* __restrict__ rewards, const float* __restrict__ dones,
        const float* __restrict__ values,  const float* __restrict__ bootval,
        float* __restrict__ returns)
{
    const int b = blockIdx.x * 256 + threadIdx.x;
    if (b >= BATCH) return;
    const float bv = bootval[b];
    float G = bv;
    #pragma unroll
    for (int t = HORIZON - 1; t >= 0; --t) {
        const float nv = (t == HORIZON - 1) ? bv : values[b * HORIZON + t + 1];
        const float m  = 1.0f - dones[b * HORIZON + t];
        G = rewards[b * HORIZON + t] +
            DISCOUNT * m * ((1.0f - LAMBDA) * nv + LAMBDA * G);
        returns[b * HORIZON + t] = G;
    }
}

// ---------------------------------------------------------------------------
// Kernel 4: per-row cross-entropy (two-hot => only 2 gathered logits) plus
// moment partials; deterministic in-block tree reduction -> 64x5 partials.
// ---------------------------------------------------------------------------
__global__ __launch_bounds__(256) void ce_stats_kernel(
        const float* __restrict__ logits, const float* __restrict__ values,
        const float* __restrict__ lse,    const float* __restrict__ returns,
        float* __restrict__ partials)
{
    __shared__ float red[5][256];
    const int r = blockIdx.x * 256 + threadIdx.x;   // 64 blocks x 256 = 16384

    const float G   = returns[r];
    const float pos = (fminf(fmaxf(G, BIN_LO), -BIN_LO) - BIN_LO) / BIN_STEP;
    int low = (int)floorf(pos);
    low = min(max(low, 0), NBINS - 2);
    const float frac = pos - (float)low;
    const float llo = logits[(size_t)r * KPAD + low];
    const float lhi = logits[(size_t)r * KPAD + low + 1];
    const float ce  = lse[r] - ((1.0f - frac) * llo + frac * lhi);
    const float v   = values[r];

    const int t = threadIdx.x;
    red[0][t] = ce;  red[1][t] = v;  red[2][t] = v * v;
    red[3][t] = G;   red[4][t] = G * G;
    __syncthreads();
    for (int s = 128; s > 0; s >>= 1) {
        if (t < s) {
            #pragma unroll
            for (int i = 0; i < 5; ++i) red[i][t] += red[i][t + s];
        }
        __syncthreads();
    }
    if (t == 0) {
        #pragma unroll
        for (int i = 0; i < 5; ++i) partials[blockIdx.x * 5 + i] = red[i][0];
    }
}

// ---------------------------------------------------------------------------
// Kernel 5: fixed-order final reduction + the 6 output scalars.
// ---------------------------------------------------------------------------
__global__ void finalize_kernel(const float* __restrict__ partials,
                                float* __restrict__ out)
{
    if (threadIdx.x != 0) return;
    float s[5] = {0.f, 0.f, 0.f, 0.f, 0.f};
    for (int b = 0; b < 64; ++b)
        for (int i = 0; i < 5; ++i) s[i] += partials[b * 5 + i];
    const float N = (float)NROWS;
    const float value_loss = s[0] / N;
    const float mean_v = s[1] / N;
    const float mean_g = s[3] / N;
    const float var_v = (s[2] - s[1] * s[1] / N) / (N - 1.0f);
    const float var_g = (s[4] - s[3] * s[3] / N) / (N - 1.0f);
    out[0] = LOSS_SCALE * value_loss;
    out[1] = value_loss;
    out[2] = mean_v;
    out[3] = mean_g;
    out[4] = sqrtf(fmaxf(var_v, 0.0f));
    out[5] = sqrtf(fmaxf(var_g, 0.0f));
}

// ---------------------------------------------------------------------------
extern "C" void kernel_launch(void* const* d_in, const int* in_sizes, int n_in,
                              void* d_out, int out_size, void* d_ws, size_t ws_size,
                              hipStream_t stream) {
    const float* latents = (const float*)d_in[0];   // (1024,16,4096)
    const float* rewards = (const float*)d_in[1];   // (1024,16)
    const float* dones   = (const float*)d_in[2];   // (1024,16)
    const float* boot    = (const float*)d_in[3];   // (1024,4096)
    const float* W       = (const float*)d_in[4];   // (4096,255)
    const float* bias    = (const float*)d_in[5];   // (255,)
    float* out = (float*)d_out;

    // Workspace layout (~19.1 MB total)
    char* ws = (char*)d_ws;
    unsigned short* packB = (unsigned short*)ws;                       // 2 MB
    size_t off = (size_t)DIM * KPAD * 2;                               // 2,097,152
    float* logits  = (float*)(ws + off); off += (size_t)NROWS * KPAD * 4;  // 16.78 MB
    float* values  = (float*)(ws + off); off += (size_t)NROWS * 4;
    float* lse     = (float*)(ws + off); off += (size_t)NROWS * 4;
    float* bootval = (float*)(ws + off); off += (size_t)BATCH * 4;
    float* returns = (float*)(ws + off); off += (size_t)NROWS * 4;
    float* partials= (float*)(ws + off);

    pack_w_kernel<<<(DIM * KPAD) / 256, 256, 0, stream>>>(W, packB);
    gemm_head_kernel<<<NBLK, 128, 0, stream>>>(latents, boot, bias, packB,
                                               logits, values, lse, bootval);
    scan_kernel<<<(BATCH + 255) / 256, 256, 0, stream>>>(rewards, dones, values,
                                                         bootval, returns);
    ce_stats_kernel<<<NROWS / 256, 256, 0, stream>>>(logits, values, lse,
                                                     returns, partials);
    finalize_kernel<<<1, 64, 0, stream>>>(partials, out);
}
